// HBGATPN_60078002536650
// MI455X (gfx1250) — matrix operand order repaired and checked by reading, not compiled, not run on hardware
//
#include <hip/hip_runtime.h>

// ---------------------------------------------------------------------------
// Types for CDNA5 WMMA (wave32): V_WMMA_F32_16X16X32_BF16
// ---------------------------------------------------------------------------
typedef __attribute__((ext_vector_type(16))) __bf16    v16bf;
typedef __attribute__((ext_vector_type(8)))  float     v8f;
typedef __attribute__((ext_vector_type(4)))  unsigned  u32x4;

static __device__ __forceinline__ __bf16 f2bf(float f) {
  // round-to-nearest-even f32 -> bf16
  unsigned u = __float_as_uint(f);
  unsigned r = (u + 0x7FFFu + ((u >> 16) & 1u)) >> 16;
  return __builtin_bit_cast(__bf16, (unsigned short)r);
}

// sortable-uint mapping so atomicMax(uint) implements float max
static __device__ __forceinline__ unsigned fsort(float f) {
  unsigned u = __float_as_uint(f);
  return (u & 0x80000000u) ? ~u : (u | 0x80000000u);
}
static __device__ __forceinline__ float funsort(unsigned s) {
  unsigned u = (s & 0x80000000u) ? (s ^ 0x80000000u) : ~s;
  return __uint_as_float(u);
}

// low 32 bits of a generic LDS pointer == LDS byte offset (ISA: LDS aperture
// keeps the offset in addr[31:0])
static __device__ __forceinline__ unsigned lds_addr32(const void* p) {
  return (unsigned)(uintptr_t)p;
}

// ---------------------------------------------------------------------------
// Node feature embedder: y = relu(layernorm(x @ W + b) * g + beta)
// one block (64 threads) per node; D = 64
// ---------------------------------------------------------------------------
__global__ void embed_kernel(const float* __restrict__ X, const float* __restrict__ W,
                             const float* __restrict__ b, const float* __restrict__ g,
                             const float* __restrict__ beta, float* __restrict__ out,
                             int N, int FD) {
  int n = blockIdx.x;
  int d = threadIdx.x;                      // 0..63
  __shared__ float sh[64];
  const float* xrow = X + (size_t)n * FD;
  float acc = b[d];
  for (int f = 0; f < FD; ++f) acc += xrow[f] * W[f * 64 + d];
  sh[d] = acc; __syncthreads();
  for (int off = 32; off > 0; off >>= 1) { if (d < off) sh[d] += sh[d + off]; __syncthreads(); }
  float mean = sh[0] * (1.0f / 64.0f);
  __syncthreads();
  float cent = acc - mean;
  sh[d] = cent * cent; __syncthreads();
  for (int off = 32; off > 0; off >>= 1) { if (d < off) sh[d] += sh[d + off]; __syncthreads(); }
  float var = sh[0] * (1.0f / 64.0f);
  float y = cent * rsqrtf(var + 1e-5f) * g[d] + beta[d];
  out[(size_t)n * 64 + d] = fmaxf(y, 0.0f);
}

// ---------------------------------------------------------------------------
// f32 -> bf16 staging conversions
// ---------------------------------------------------------------------------
__global__ void f32_to_bf16_kernel(const float* __restrict__ src, __bf16* __restrict__ dst, int n) {
  int i = blockIdx.x * blockDim.x + threadIdx.x;
  if (i < n) dst[i] = f2bf(src[i]);
}

// W [64,256] f32 (row-major) -> Wb [256,64] bf16 (transposed: K contiguous per column)
__global__ void w_transpose_bf16_kernel(const float* __restrict__ W, __bf16* __restrict__ Wb) {
  int i = blockIdx.x * blockDim.x + threadIdx.x;   // 0 .. 64*256-1
  if (i >= 64 * 256) return;
  int k = i >> 8, n = i & 255;
  Wb[(size_t)n * 64 + k] = f2bf(W[i]);
}

// ---------------------------------------------------------------------------
// P[N,256] = Hb[N,64] @ W[64,256]  (W given as WbT[256,64] bf16)
//
// Block = 256 threads (8 waves); block computes a 16-row strip across all 256
// output columns. The full B panel (32 KB) + the 16x64 A strip (2 KB) are
// staged into LDS with gfx1250 async global->LDS b128 copies (ASYNCcnt +
// s_wait_asynccnt), then each wave computes two 16x16 tiles (4 WMMAs) with
// ds_load_b128 fragment reads. K=64 -> two 16x16x32 bf16 WMMA steps per tile.
// ---------------------------------------------------------------------------
__global__ __launch_bounds__(256) void gemm64x256_wmma_lds(
    const __bf16* __restrict__ Hb, const __bf16* __restrict__ WbT,
    float* __restrict__ P, int N) {
  __shared__ __align__(16) __bf16 Bsh[256 * 64];   // 32 KB: entire WbT panel
  __shared__ __align__(16) __bf16 Ash[16 * 64];    // 2 KB: 16-row A strip

  int tid = threadIdx.x;
  int m0 = blockIdx.x * 16;

  // --- async-stage B panel: each thread copies one 128B column (8 x b128) ---
  {
    const __bf16* g = WbT + (size_t)tid * 64;
    unsigned l = lds_addr32(Bsh + tid * 64);
#pragma unroll
    for (int j = 0; j < 8; ++j) {
      asm volatile("global_load_async_to_lds_b128 %0, %1, off"
                   :: "v"(l + j * 16), "v"(g + j * 8) : "memory");
    }
  }
  // --- async-stage A strip: threads 0..127 copy 16B each (rows clamped) ---
  if (tid < 128) {
    int r = tid >> 3, j8 = tid & 7;
    int row = m0 + r;
    int rc = (row < N) ? row : (N - 1);
    const __bf16* g = Hb + (size_t)rc * 64 + j8 * 8;
    unsigned l = lds_addr32(Ash + r * 64 + j8 * 8);
    asm volatile("global_load_async_to_lds_b128 %0, %1, off"
                 :: "v"(l), "v"(g) : "memory");
  }
  asm volatile("s_wait_asynccnt 0x0" ::: "memory");
  __syncthreads();

  int lane = tid & 31;
  int wave = tid >> 5;
  int l15 = lane & 15;
  int hi  = lane >> 4;
  int row = m0 + l15;
  unsigned msk = (row < N) ? 0xFFFFFFFFu : 0u;
  u32x4 mv = {msk, msk, msk, msk};

  const __bf16* arow = Ash + l15 * 64;
  const __bf16* bc0  = Bsh + (size_t)(wave * 32 + l15) * 64;   // col tile 2w
  const __bf16* bc1  = bc0 + 16 * 64;                          // col tile 2w+1

  v8f c0 = {0.f, 0.f, 0.f, 0.f, 0.f, 0.f, 0.f, 0.f};
  v8f c1 = {0.f, 0.f, 0.f, 0.f, 0.f, 0.f, 0.f, 0.f};
#pragma unroll
  for (int kk = 0; kk < 64; kk += 32) {
    // A 16x32 bf16 layout: lanes 0-15 hold K kk+0..15; lanes 16-31 K kk+16..31 (sequential)
    int kbA = kk + hi * 16;
    union { u32x4 u[2]; v16bf v; } A, B0, B1;
    A.u[0] = *(const u32x4*)(arow + kbA)      & mv;
    A.u[1] = *(const u32x4*)(arow + kbA + 8) & mv;
    // B 32x16 bf16 layout: lanes 0-15 hold K kk+0..15; lanes 16-31 K kk+16..31 (sequential)
    int kbB = kk + hi * 16;
    B0.u[0] = *(const u32x4*)(bc0 + kbB);
    B0.u[1] = *(const u32x4*)(bc0 + kbB + 8);
    B1.u[0] = *(const u32x4*)(bc1 + kbB);
    B1.u[1] = *(const u32x4*)(bc1 + kbB + 8);
    c0 = __builtin_amdgcn_wmma_f32_16x16x32_bf16(false, A.v, false, B0.v, (short)0, c0, false, false);
    c1 = __builtin_amdgcn_wmma_f32_16x16x32_bf16(false, A.v, false, B1.v, (short)0, c1, false, false);
  }

  // C/D layout: VGPR r -> M = m0 + r + hi*8, N = col
  int col0 = wave * 32 + l15;
  if (m0 + 16 <= N) {
#pragma unroll
    for (int r = 0; r < 8; ++r) {
      size_t ro = (size_t)(m0 + r + hi * 8) * 256;
      P[ro + col0]      = c0[r];
      P[ro + col0 + 16] = c1[r];
    }
  } else {
#pragma unroll
    for (int r = 0; r < 8; ++r) {
      int rr = m0 + r + hi * 8;
      if (rr < N) {
        P[(size_t)rr * 256 + col0]      = c0[r];
        P[(size_t)rr * 256 + col0 + 16] = c1[r];
      }
    }
  }
}

// ---------------------------------------------------------------------------
// Edge phase: GATv2 logits + segment softmax via L2 atomics
// ---------------------------------------------------------------------------
__global__ void edge_logits_kernel(const float* __restrict__ xl, const float* __restrict__ xr,
                                   const int* __restrict__ ei, const float* __restrict__ att,
                                   float* __restrict__ logits, unsigned* __restrict__ msort, int E) {
  int idx = blockIdx.x * blockDim.x + threadIdx.x;
  if (idx >= E * 4) return;
  int e = idx >> 2, h = idx & 3;
  int src = ei[e], dst = ei[E + e];
  const float* pl = xl + (size_t)src * 256 + h * 64;
  const float* pr = xr + (size_t)dst * 256 + h * 64;
  const float* pa = att + h * 64;
  float s = 0.0f;
#pragma unroll 8
  for (int d = 0; d < 64; ++d) {
    float v = pl[d] + pr[d];
    v = (v > 0.0f) ? v : 0.2f * v;              // leaky_relu(0.2)
    s += v * pa[d];
  }
  logits[idx] = s;
  atomicMax(&msort[(size_t)dst * 4 + h], fsort(s));
}

__global__ void edge_exp_kernel(const float* __restrict__ logits, const int* __restrict__ ei,
                                const unsigned* __restrict__ msort, float* __restrict__ exb,
                                float* __restrict__ denom, int E) {
  int idx = blockIdx.x * blockDim.x + threadIdx.x;
  if (idx >= E * 4) return;
  int e = idx >> 2, h = idx & 3;
  int dst = ei[E + e];
  float m  = funsort(msort[(size_t)dst * 4 + h]);
  float ex = __expf(logits[idx] - m);
  exb[idx] = ex;
  atomicAdd(&denom[(size_t)dst * 4 + h], ex);
}

// head-mean folded in: agg[dst,d] += 0.25 * sum_h alpha[e,h] * xl[src, h*64+d]
__global__ void edge_scatter_kernel(const float* __restrict__ xl, const int* __restrict__ ei,
                                    const float* __restrict__ exb, const float* __restrict__ denom,
                                    float* __restrict__ agg, int E) {
  int idx = blockIdx.x * blockDim.x + threadIdx.x;
  if (idx >= E * 64) return;
  int e = idx >> 6, d = idx & 63;
  int src = ei[e], dst = ei[E + e];
  float acc = 0.0f;
#pragma unroll
  for (int h = 0; h < 4; ++h) {
    float alpha = exb[(size_t)e * 4 + h] / denom[(size_t)dst * 4 + h];
    acc += alpha * xl[(size_t)src * 256 + h * 64 + d];
  }
  atomicAdd(&agg[(size_t)dst * 64 + d], 0.25f * acc);
}

// x = relu(agg + sum(bias)) + x   (HeteroConv sum aggr + ReLU + residual)
__global__ void combine_kernel(float* __restrict__ x, const float* __restrict__ agg,
                               const float* b0, const float* b1, const float* b2, int N) {
  int idx = blockIdx.x * blockDim.x + threadIdx.x;
  if (idx >= N * 64) return;
  int d = idx & 63;
  float bs = 0.0f;
  if (b0) bs += b0[d];
  if (b1) bs += b1[d];
  if (b2) bs += b2[d];
  float v = agg[idx] + bs;
  x[idx] = fmaxf(v, 0.0f) + x[idx];
}

// per-channel mean + max over nodes; blockIdx.x = channel d (64 blocks)
__global__ void readout_kernel(const float* __restrict__ x, int N,
                               float* __restrict__ om, float* __restrict__ ox) {
  int d = blockIdx.x;
  __shared__ float ss[256];
  __shared__ float sm[256];
  float s = 0.0f, mx = -3.402823466e38f;
  for (int n = threadIdx.x; n < N; n += blockDim.x) {
    float v = x[(size_t)n * 64 + d];
    s += v;
    mx = fmaxf(mx, v);
  }
  ss[threadIdx.x] = s; sm[threadIdx.x] = mx; __syncthreads();
  for (int off = 128; off > 0; off >>= 1) {
    if ((int)threadIdx.x < off) {
      ss[threadIdx.x] += ss[threadIdx.x + off];
      sm[threadIdx.x] = fmaxf(sm[threadIdx.x], sm[threadIdx.x + off]);
    }
    __syncthreads();
  }
  if (threadIdx.x == 0) { om[d] = ss[0] / (float)N; ox[d] = sm[0]; }
}

// ---------------------------------------------------------------------------
// Host orchestration
// ---------------------------------------------------------------------------
extern "C" void kernel_launch(void* const* d_in, const int* in_sizes, int n_in,
                              void* d_out, int out_size, void* d_ws, size_t ws_size,
                              hipStream_t stream) {
  const float* x_in[3] = {(const float*)d_in[0], (const float*)d_in[1], (const float*)d_in[2]};
  const int* ei[5];
  int Ecnt[5];
  for (int i = 0; i < 5; ++i) { ei[i] = (const int*)d_in[3 + i]; Ecnt[i] = in_sizes[3 + i] / 2; }
  const float* Wemb[3]    = {(const float*)d_in[8],  (const float*)d_in[12], (const float*)d_in[16]};
  const float* bemb[3]    = {(const float*)d_in[9],  (const float*)d_in[13], (const float*)d_in[17]};
  const float* gemb[3]    = {(const float*)d_in[10], (const float*)d_in[14], (const float*)d_in[18]};
  const float* betaemb[3] = {(const float*)d_in[11], (const float*)d_in[15], (const float*)d_in[19]};
  const float* gat_Wl   = (const float*)d_in[20];
  const float* gat_Wr   = (const float*)d_in[21];
  const float* gat_att  = (const float*)d_in[22];
  const float* gat_bias = (const float*)d_in[23];

  int FD[3] = {16, 12, 10};
  int Nn[3] = {in_sizes[0] / 16, in_sizes[1] / 12, in_sizes[2] / 10};

  // output regions double as the live node-state buffers
  float* x_t = (float*)d_out;
  float* x_w = x_t + (size_t)Nn[0] * 64;
  float* x_s = x_w + (size_t)Nn[1] * 64;
  float* gc  = x_s + (size_t)Nn[2] * 64;
  float* xb[3] = {x_t, x_w, x_s};

  // workspace carve (256B-aligned chunks)
  int Emax = 0; for (int i = 0; i < 5; ++i) if (Ecnt[i] > Emax) Emax = Ecnt[i];
  char* wsbase = (char*)d_ws;
  size_t wsoff = 0;
  auto walloc = [&](size_t bytes) -> void* {
    void* r = wsbase + wsoff;
    wsoff = (wsoff + bytes + 255) & ~(size_t)255;
    return r;
  };
  float*    xl     = (float*)   walloc((size_t)Nn[0] * 256 * sizeof(float));
  float*    xr     = (float*)   walloc((size_t)Nn[0] * 256 * sizeof(float));
  float*    logits = (float*)   walloc((size_t)Emax * 4 * sizeof(float));
  float*    exb    = (float*)   walloc((size_t)Emax * 4 * sizeof(float));
  unsigned* msort  = (unsigned*)walloc((size_t)Nn[0] * 4 * sizeof(unsigned));
  float*    denom  = (float*)   walloc((size_t)Nn[0] * 4 * sizeof(float));
  float*    aggb[3];
  for (int i = 0; i < 3; ++i) aggb[i] = (float*)walloc((size_t)Nn[i] * 64 * sizeof(float));
  __bf16* hb[3];
  for (int i = 0; i < 3; ++i) hb[i] = (__bf16*)walloc((size_t)Nn[i] * 64 * sizeof(__bf16));
  __bf16* wbl = (__bf16*)walloc((size_t)256 * 64 * sizeof(__bf16));
  __bf16* wbr = (__bf16*)walloc((size_t)256 * 64 * sizeof(__bf16));

  // 1) per-type feature embedder
  for (int i = 0; i < 3; ++i)
    embed_kernel<<<Nn[i], 64, 0, stream>>>(x_in[i], Wemb[i], bemb[i], gemb[i], betaemb[i],
                                           xb[i], Nn[i], FD[i]);

  // edge type -> (src type, dst type): 0=task 1=worker 2=station
  const int ST[5] = {0, 0, 2, 1, 0};
  const int DT[5] = {0, 2, 0, 0, 1};

  for (int l = 0; l < 2; ++l) {
    // stage current node states as bf16 (once per layer per type)
    for (int i = 0; i < 3; ++i) {
      int n = Nn[i] * 64;
      f32_to_bf16_kernel<<<(n + 255) / 256, 256, 0, stream>>>(xb[i], hb[i], n);
      hipMemsetAsync(aggb[i], 0, (size_t)Nn[i] * 64 * sizeof(float), stream);
    }

    for (int t = 0; t < 5; ++t) {
      int s = ST[t], dtp = DT[t];
      int Ns = Nn[s], Nd = Nn[dtp], E = Ecnt[t];
      const float* Wl  = gat_Wl  + (size_t)(l * 5 + t) * 64 * 256;
      const float* Wr  = gat_Wr  + (size_t)(l * 5 + t) * 64 * 256;
      const float* att = gat_att + (size_t)(l * 5 + t) * 4 * 64;

      // stage weights as transposed bf16 [256,64]
      w_transpose_bf16_kernel<<<(64 * 256 + 255) / 256, 256, 0, stream>>>(Wl, wbl);
      w_transpose_bf16_kernel<<<(64 * 256 + 255) / 256, 256, 0, stream>>>(Wr, wbr);

      gemm64x256_wmma_lds<<<(Ns + 15) / 16, 256, 0, stream>>>(hb[s],   wbl, xl, Ns);
      gemm64x256_wmma_lds<<<(Nd + 15) / 16, 256, 0, stream>>>(hb[dtp], wbr, xr, Nd);

      hipMemsetAsync(msort, 0, (size_t)Nd * 4 * sizeof(unsigned), stream);
      hipMemsetAsync(denom, 0, (size_t)Nd * 4 * sizeof(float), stream);

      int nth = E * 4;
      edge_logits_kernel<<<(nth + 255) / 256, 256, 0, stream>>>(xl, xr, ei[t], att, logits, msort, E);
      edge_exp_kernel<<<(nth + 255) / 256, 256, 0, stream>>>(logits, ei[t], msort, exb, denom, E);
      int nsc = E * 64;
      edge_scatter_kernel<<<(nsc + 255) / 256, 256, 0, stream>>>(xl, ei[t], exb, denom, aggb[dtp], E);
    }

    // biases: task <- t{0,2,3}; station <- t1; worker <- t4
    const float* bt0 = gat_bias + (size_t)(l * 5 + 0) * 64;
    const float* bs1 = gat_bias + (size_t)(l * 5 + 1) * 64;
    const float* bt2 = gat_bias + (size_t)(l * 5 + 2) * 64;
    const float* bt3 = gat_bias + (size_t)(l * 5 + 3) * 64;
    const float* bw4 = gat_bias + (size_t)(l * 5 + 4) * 64;
    combine_kernel<<<(Nn[0] * 64 + 255) / 256, 256, 0, stream>>>(x_t, aggb[0], bt0, bt2, bt3, Nn[0]);
    combine_kernel<<<(Nn[1] * 64 + 255) / 256, 256, 0, stream>>>(x_w, aggb[1], bw4, nullptr, nullptr, Nn[1]);
    combine_kernel<<<(Nn[2] * 64 + 255) / 256, 256, 0, stream>>>(x_s, aggb[2], bs1, nullptr, nullptr, Nn[2]);
  }

  // global context: [station.mean, task.mean, worker.mean, station.max, task.max, worker.max]
  readout_kernel<<<64, 256, 0, stream>>>(x_s, Nn[2], gc + 0,   gc + 192);
  readout_kernel<<<64, 256, 0, stream>>>(x_t, Nn[0], gc + 64,  gc + 256);
  readout_kernel<<<64, 256, 0, stream>>>(x_w, Nn[1], gc + 128, gc + 320);
}
